// Att_3513283248172
// MI455X (gfx1250) — matrix operand order repaired and checked by reading, not compiled, not run on hardware
//
#include <hip/hip_runtime.h>

// MI455X (gfx1250) fused additive-attention:
//   K0: pack W (f32->bf16) into WMMA B-fragment layout
//   K1: mask + bf16 WMMA GEMM (gru @ W) + tanh + dot(u) + mask-apply -> scores
//       512-thread WG (16 waves -> ~256 VGPR budget, no spills),
//       3-stage B pipeline (global/L2) + 2-stage A pipeline (LDS),
//       pinned with sched_barrier so the backend cannot collapse the buffers.
//   K2: softmax over S per batch
//   K3: weighted sum over S -> [B, D]

typedef __attribute__((ext_vector_type(16))) __bf16 v16bf;
typedef __attribute__((ext_vector_type(4)))  __bf16 v4bf;
typedef __attribute__((ext_vector_type(8)))  float  v8f;

union Frag32 { v16bf v; float4 f4[2]; };

#define B_DIM 32
#define S_DIM 2048
#define D_DIM 1024
#define A_DIM 512
#define ROWS_PER_WG 32          // M rows per workgroup (2 x 16-row subtiles)
#define KB_COUNT (D_DIM / 32)   // 32 K-blocks of 32
#define NT_COUNT (A_DIM / 16)   // 32 N-tiles of 16

// ---------------- K0: pack W into per-lane B fragments ----------------
// Wpack[((kb*NT + nt)*32 + lane)*16 + i] = bf16( W[(kb*32 + (lane<16?0:16) + i)*A + nt*16 + (lane&15)] )
__global__ void pack_w_kernel(const float* __restrict__ w, __bf16* __restrict__ wpack) {
    int e = blockIdx.x * blockDim.x + threadIdx.x;  // 0 .. 524287
    int i    = e & 15;
    int lane = (e >> 4) & 31;
    int nt   = (e >> 9) & 31;
    int kb   = e >> 14;
    int K = kb * 32 + ((lane < 16) ? 0 : 16) + i;
    int N = nt * 16 + (lane & 15);
    wpack[e] = (__bf16)w[K * A_DIM + N];
}

// ---------------- K1: mask + GEMM + tanh + dot(u) -> scores ----------------
__global__ __launch_bounds__(512, 1)
__attribute__((amdgpu_waves_per_eu(4, 8)))
void score_kernel(const float* __restrict__ x,
                  const float* __restrict__ gru,
                  const __bf16* __restrict__ wpack,
                  const float* __restrict__ u,
                  float* __restrict__ scores) {
    __shared__ __align__(16) __bf16 Alds[ROWS_PER_WG * D_DIM];  // 64 KB
    __shared__ float s_scores[ROWS_PER_WG];
    __shared__ float s_mask[ROWS_PER_WG];

    const int tid  = threadIdx.x;        // 0..511
    const int lane = tid & 31;
    const int wave = tid >> 5;           // 0..15
    const size_t m0 = (size_t)blockIdx.x * ROWS_PER_WG;

    // Stage 32 gru rows into LDS as bf16: float4 loads, 8-byte bf16x4 stores.
    {
        const int rowi = tid >> 8;       // 0..1
        const int colv = tid & 255;      // float4 index within row
        #pragma unroll
        for (int it = 0; it < 16; ++it) {
            const int row = it * 2 + rowi;
            float4 v = *(const float4*)&gru[(m0 + row) * D_DIM + colv * 4];
            v4bf h;
            h.x = (__bf16)v.x; h.y = (__bf16)v.y; h.z = (__bf16)v.z; h.w = (__bf16)v.w;
            *(v4bf*)&Alds[row * D_DIM + colv * 4] = h;
        }
    }

    // Mask: wave w sums x rows 2w and 2w+1 (sign(|sum|): 0 iff sum == 0)
    #pragma unroll
    for (int rr = 0; rr < 2; ++rr) {
        const int row = 2 * wave + rr;
        const float* xr = x + (m0 + row) * D_DIM;
        float s = 0.0f;
        #pragma unroll
        for (int j = 0; j < 8; ++j) {
            float4 v = *(const float4*)&xr[(j * 32 + lane) * 4];
            s += (v.x + v.y) + (v.z + v.w);
        }
        #pragma unroll
        for (int off = 16; off >= 1; off >>= 1) s += __shfl_xor(s, off, 32);
        if (lane == 0) s_mask[row] = s;
    }
    if (tid < ROWS_PER_WG) s_scores[tid] = 0.0f;
    __syncthreads();

    // Wave tiling: mi in 0..1 (16-row subtile), ni in 0..7 (64-column slab = 4 N-tiles)
    const int mi    = wave >> 3;
    const int ni    = wave & 7;
    const int nlane = lane & 15;
    // A k order used here (contiguous, same as Wpack B): lanes 0-15 hold K {0..15}; lanes 16-31 hold K {16..31}
    const int kofs = (lane < 16) ? 0 : 16;
    const __bf16* arow = &Alds[(16 * mi + nlane) * D_DIM];
    const __bf16* bwave = wpack + (((size_t)ni * 4 * 32) + lane) * 16;
    // per-kb stride in Wpack: NT_COUNT tiles * 32 lanes * 16 bf16
    const size_t kbstride = (size_t)NT_COUNT * 32 * 16;

    v8f c0 = {}, c1 = {}, c2 = {}, c3 = {};

    Frag32 a[2];        // 2-stage A pipeline (LDS)
    Frag32 b[3][4];     // 3-stage B pipeline (global/L2)

    auto loadA = [&](int kb, int buf) {
        a[buf].f4[0] = *(const float4*)(arow + kb * 32 + kofs);
        a[buf].f4[1] = *(const float4*)(arow + kb * 32 + kofs + 8);
    };
    auto loadB = [&](int kb, int buf) {
        const __bf16* bb = bwave + (size_t)kb * kbstride;
        #pragma unroll
        for (int t = 0; t < 4; ++t) {
            b[buf][t].f4[0] = ((const float4*)(bb + t * 512))[0];
            b[buf][t].f4[1] = ((const float4*)(bb + t * 512))[1];
        }
    };

    // Prologue: B two stages deep, A one stage deep.
    loadB(0, 0);
    loadB(1, 1);
    loadA(0, 0);
    #pragma unroll
    for (int kb = 0; kb < KB_COUNT; ++kb) {
        // Prefetch stage kb+2 (B) into the buffer freed by iteration kb-1,
        // and stage kb+1 (A). Issued BEFORE the WMMA group of kb.
        if (kb + 2 < KB_COUNT) loadB(kb + 2, (kb + 2) % 3);
        if (kb + 1 < KB_COUNT) loadA(kb + 1, (kb + 1) & 1);
        __builtin_amdgcn_sched_barrier(0);   // loads may not sink past here
        {
            const int ca = kb & 1;
            const int cb = kb % 3;
            c0 = __builtin_amdgcn_wmma_f32_16x16x32_bf16(false, a[ca].v, false, b[cb][0].v, (short)0, c0, false, false);
            c1 = __builtin_amdgcn_wmma_f32_16x16x32_bf16(false, a[ca].v, false, b[cb][1].v, (short)0, c1, false, false);
            c2 = __builtin_amdgcn_wmma_f32_16x16x32_bf16(false, a[ca].v, false, b[cb][2].v, (short)0, c2, false, false);
            c3 = __builtin_amdgcn_wmma_f32_16x16x32_bf16(false, a[ca].v, false, b[cb][3].v, (short)0, c3, false, false);
        }
        __builtin_amdgcn_sched_barrier(0);   // wmmas may not sink into next stage's loads
    }

    // Epilogue: score partial = sum_n tanh(C[m,n]) * u[n] over this wave's 64 columns.
    // C layout: VGPR r, lanes 0-15 -> M=r, N=lane; lanes 16-31 -> M=8+r, N=lane-16.
    const float u0 = u[ni * 64 +      nlane];
    const float u1 = u[ni * 64 + 16 + nlane];
    const float u2 = u[ni * 64 + 32 + nlane];
    const float u3 = u[ni * 64 + 48 + nlane];
    const int rowbase = 16 * mi + ((lane < 16) ? 0 : 8);
    #pragma unroll
    for (int r = 0; r < 8; ++r) {
        float acc = tanhf(c0[r]) * u0 + tanhf(c1[r]) * u1
                  + tanhf(c2[r]) * u2 + tanhf(c3[r]) * u3;
        // reduce across the 16 lanes of this half (same row, different columns)
        #pragma unroll
        for (int off = 1; off < 16; off <<= 1) acc += __shfl_xor(acc, off, 32);
        if (nlane == 0) atomicAdd(&s_scores[rowbase + r], acc);
    }
    __syncthreads();

    if (tid < ROWS_PER_WG) {
        float sc = s_scores[tid];
        sc = (s_mask[tid] != 0.0f) ? sc : -1.0e9f;
        scores[m0 + tid] = sc;
    }
}

// ---------------- K2: softmax over S per batch ----------------
__global__ __launch_bounds__(256)
void softmax_kernel(const float* __restrict__ sc, float* __restrict__ al) {
    __shared__ float red[256];
    const int b = blockIdx.x;
    const int t = threadIdx.x;
    const float* row = sc + (size_t)b * S_DIM;

    float v[8];
    float m = -3.4e38f;
    #pragma unroll
    for (int i = 0; i < 8; ++i) { v[i] = row[t + i * 256]; m = fmaxf(m, v[i]); }
    red[t] = m; __syncthreads();
    for (int s = 128; s > 0; s >>= 1) { if (t < s) red[t] = fmaxf(red[t], red[t + s]); __syncthreads(); }
    m = red[0]; __syncthreads();

    float s = 0.0f;
    #pragma unroll
    for (int i = 0; i < 8; ++i) { v[i] = expf(v[i] - m); s += v[i]; }
    red[t] = s; __syncthreads();
    for (int st = 128; st > 0; st >>= 1) { if (t < st) red[t] += red[t + st]; __syncthreads(); }
    const float inv = 1.0f / red[0];
    #pragma unroll
    for (int i = 0; i < 8; ++i) al[(size_t)b * S_DIM + t + i * 256] = v[i] * inv;
}

// ---------------- K3: out[b,d] = sum_s gru[b,s,d] * alpha[b,s] ----------------
__global__ __launch_bounds__(256)
void wsum_kernel(const float* __restrict__ gru, const float* __restrict__ al,
                 float* __restrict__ out) {
    const int b = blockIdx.y;
    const int d = blockIdx.x * 256 + threadIdx.x;
    const float* g = gru + (size_t)b * S_DIM * D_DIM + d;
    const float* a = al + (size_t)b * S_DIM;
    float acc0 = 0.0f, acc1 = 0.0f, acc2 = 0.0f, acc3 = 0.0f;
    #pragma unroll 2
    for (int s = 0; s < S_DIM; s += 4) {
        acc0 += g[(size_t)(s + 0) * D_DIM] * a[s + 0];
        acc1 += g[(size_t)(s + 1) * D_DIM] * a[s + 1];
        acc2 += g[(size_t)(s + 2) * D_DIM] * a[s + 2];
        acc3 += g[(size_t)(s + 3) * D_DIM] * a[s + 3];
    }
    out[(size_t)b * D_DIM + d] = (acc0 + acc1) + (acc2 + acc3);
}

extern "C" void kernel_launch(void* const* d_in, const int* in_sizes, int n_in,
                              void* d_out, int out_size, void* d_ws, size_t ws_size,
                              hipStream_t stream) {
    const float* x   = (const float*)d_in[0];   // [B, S, D]
    const float* gru = (const float*)d_in[1];   // [B, S, D]
    const float* w   = (const float*)d_in[2];   // [D, A]
    const float* u   = (const float*)d_in[3];   // [A]
    float* out = (float*)d_out;                 // [B, D]

    // Workspace layout: Wpack (1 MB) | scores (256 KB) | alphas (256 KB)
    __bf16* wpack  = (__bf16*)d_ws;
    float*  scores = (float*)((char*)d_ws + (1u << 20));
    float*  alphas = (float*)((char*)d_ws + (1u << 20) + (256u << 10));

    // K0: pack W (D*A = 524288 elements)
    pack_w_kernel<<<(D_DIM * A_DIM) / 256, 256, 0, stream>>>(w, wpack);

    // K1: scores. (B*S)/32 = 2048 blocks of 512 threads.
    score_kernel<<<(B_DIM * S_DIM) / ROWS_PER_WG, 512, 0, stream>>>(x, gru, wpack, u, scores);

    // K2: softmax per batch
    softmax_kernel<<<B_DIM, 256, 0, stream>>>(scores, alphas);

    // K3: weighted sum
    wsum_kernel<<<dim3(D_DIM / 256, B_DIM), 256, 0, stream>>>(gru, alphas, out);
}
